// TransformerBlock_37838661877849
// MI455X (gfx1250) — matrix operand order repaired and checked by reading, not compiled, not run on hardware
//
#include <hip/hip_runtime.h>
#include <stdint.h>

// ---------------------------------------------------------------------------
// Transformer encoder block for MI455X (gfx1250, wave32).
// bf16 WMMA (v_wmma_f32_16x16x32_bf16) with f32 accumulation; double-buffered
// LDS staging driven by CDNA5 async global->LDS copies (ASYNCcnt).
// ---------------------------------------------------------------------------

#define D_MODEL 1024
#define N_HEADS 16
#define D_K     64
#define D_FF    4096
#define NTOK    4096   // B*S = 4*1024
#define SEQ     1024

typedef __attribute__((ext_vector_type(16))) __bf16 v16bf;
typedef __attribute__((ext_vector_type(8)))  __bf16 v8bf;
typedef __attribute__((ext_vector_type(8)))  float  v8f;
typedef __attribute__((ext_vector_type(4)))  int    v4i;

#define WMMA_BF16(a, b, c) \
  __builtin_amdgcn_wmma_f32_16x16x32_bf16(false, (a), false, (b), (short)0, (c), false, false)

// Pointer types matching the async-to-LDS builtin signature:
// (v4i __global*, v4i __shared*, imm offset, imm cpol)
typedef __attribute__((address_space(1))) v4i glob_v4i;
typedef __attribute__((address_space(3))) v4i lds_v4i;

// 16-byte global -> LDS copy on the CDNA5 async path (tracked by ASYNCcnt).
__device__ __forceinline__ void cp16_g2l(const void* g, void* l) {
#if __has_builtin(__builtin_amdgcn_global_load_async_to_lds_b128)
  __builtin_amdgcn_global_load_async_to_lds_b128(
      (glob_v4i*)(uintptr_t)g,
      (lds_v4i*)(uint32_t)(uintptr_t)l,
      0, 0);
#else
  *(v8bf*)l = *(const v8bf*)g;
#endif
}

__device__ __forceinline__ void async_wait_all() {
#if __has_builtin(__builtin_amdgcn_s_wait_asynccnt)
  __builtin_amdgcn_s_wait_asynccnt(0);
#endif
}

// Build a 16-element bf16 fragment from two 16-byte chunks (two ds/global b128s).
__device__ __forceinline__ v16bf ld_frag(const __bf16* p0, const __bf16* p1) {
  v8bf a = *(const v8bf*)p0;
  v8bf b = *(const v8bf*)p1;
  v16bf r;
#pragma unroll
  for (int i = 0; i < 8; ++i) { r[i] = a[i]; r[i + 8] = b[i]; }
  return r;
}

// ---------------------------------------------------------------------------
// Generic bf16 WMMA GEMM: C[M,N] = A[M,K] * B[K,N] + bias, optional ReLU.
// BM=BN=128, BK=32; 256 threads = 8 waves, each wave owns a 32x64 sub-tile
// (2x4 grid of 16x16 f32 accumulators). Double-buffered LDS pipeline:
//   frag-loads(cur) -> async A(nxt) + global B(nxt) -> 8x WMMA -> B stores(nxt)
//   -> s_wait_asynccnt 0 -> barrier.
// ---------------------------------------------------------------------------
#define BM  128
#define BN  128
#define BK  32
#define BKP 40   // padded K-stride (elements) for transposed B tile: 80B rows, 16B aligned

__global__ __launch_bounds__(256) void wmma_gemm_bf16(
    const __bf16* __restrict__ A, int lda,
    const __bf16* __restrict__ B, int ldb,
    const float* __restrict__ bias,
    float* __restrict__ Cf, __bf16* __restrict__ Ch, int ldc,
    int K, int relu)
{
  __shared__ alignas(128) __bf16 As[2][BM * BK];    // 2 x 8 KB
  __shared__ alignas(128) __bf16 Bs[2][BN * BKP];   // 2 x 10 KB (transposed, padded)

  const int tid  = threadIdx.x;
  const int lane = tid & 31;
  const int wid  = tid >> 5;
  const int wm   = (wid >> 1) * 32;   // wave row offset within block tile
  const int wn   = (wid & 1) * 64;    // wave col offset within block tile
  const int bm   = blockIdx.y * BM;
  const int bn   = blockIdx.x * BN;
  const int lm   = lane & 15;
  const int lh   = lane >> 4;

  // Per-thread staging coordinates.
  const int ar0 = tid >> 1;                 // A rows handled by this thread
  const int aq0 = (tid & 1) * 2;            // 2 consecutive 16B chunks
  const int bkk = tid >> 4;                 // B source row (k) 0..15 (+16 for 2nd)
  const int bn8 = tid & 15;                 // B 8-column group

  v8f acc[2][4] = {};

  // ---- prologue: stage k-tile 0 into buffer 0 -----------------------------
  {
#pragma unroll
    for (int q = 0; q < 2; ++q)
      cp16_g2l(A + (size_t)(bm + ar0) * lda + (aq0 + q) * 8,
               &As[0][ar0 * BK + (aq0 + q) * 8]);
    v8bf r0 = *(const v8bf*)(B + (size_t)(bkk)      * ldb + bn + bn8 * 8);
    v8bf r1 = *(const v8bf*)(B + (size_t)(bkk + 16) * ldb + bn + bn8 * 8);
#pragma unroll
    for (int e = 0; e < 8; ++e) Bs[0][(bn8 * 8 + e) * BKP + bkk]      = r0[e];
#pragma unroll
    for (int e = 0; e < 8; ++e) Bs[0][(bn8 * 8 + e) * BKP + bkk + 16] = r1[e];
  }
  async_wait_all();
  __syncthreads();

  const int nk = K / BK;
  for (int kt = 0; kt < nk; ++kt) {
    const int cur = kt & 1;
    const int nxt = cur ^ 1;
    const bool have_next = (kt + 1) < nk;
    const int k1 = (kt + 1) * BK;

    // Fragment loads from current buffer (issued first; LDS is in-order per
    // wave, so these are not blocked by the staging stores below).
    // A: lanes 0-15 hold K{0..7,16..23}, lanes 16-31 hold K{8..15,24..31}.
    // B: lane l = column (l&15), same K split as A: K{0..7,16..23} / K{8..15,24..31}.
    v16bf af[2], bf[4];
#pragma unroll
    for (int mt = 0; mt < 2; ++mt) {
      const __bf16* p = &As[cur][(wm + mt * 16 + lm) * BK + lh * 8];
      af[mt] = ld_frag(p, p + 16);
    }
#pragma unroll
    for (int nt = 0; nt < 4; ++nt) {
      const __bf16* p = &Bs[cur][(wn + nt * 16 + lm) * BKP + lh * 8];
      bf[nt] = ld_frag(p, p + 16);
    }

    // Kick off next tile's data movement: async DMA for A, global loads for B.
    v8bf r0{}, r1{};
    if (have_next) {
#pragma unroll
      for (int q = 0; q < 2; ++q)
        cp16_g2l(A + (size_t)(bm + ar0) * lda + k1 + (aq0 + q) * 8,
                 &As[nxt][ar0 * BK + (aq0 + q) * 8]);
      r0 = *(const v8bf*)(B + (size_t)(k1 + bkk)      * ldb + bn + bn8 * 8);
      r1 = *(const v8bf*)(B + (size_t)(k1 + bkk + 16) * ldb + bn + bn8 * 8);
      if (kt + 2 < nk) {  // pull k-tile+2 toward L2 (global_prefetch_b8)
        __builtin_prefetch(A + (size_t)(bm + ar0) * lda + k1 + BK, 0, 1);
        __builtin_prefetch(B + (size_t)(k1 + BK + bkk) * ldb + bn + bn8 * 8, 0, 1);
      }
    }

    // Matrix math covers the global-load latency of the staging above.
#pragma unroll
    for (int mt = 0; mt < 2; ++mt)
#pragma unroll
      for (int nt = 0; nt < 4; ++nt)
        acc[mt][nt] = WMMA_BF16(af[mt], bf[nt], acc[mt][nt]);

    // Transpose-store B into the next buffer (loadcnt wait lands here, after
    // the WMMAs have been issued).
    if (have_next) {
#pragma unroll
      for (int e = 0; e < 8; ++e) Bs[nxt][(bn8 * 8 + e) * BKP + bkk]      = r0[e];
#pragma unroll
      for (int e = 0; e < 8; ++e) Bs[nxt][(bn8 * 8 + e) * BKP + bkk + 16] = r1[e];
      async_wait_all();
    }
    __syncthreads();
  }

  // Epilogue: C layout = VGPR j -> row (lane<16 ? j : j+8), col lane%16.
#pragma unroll
  for (int mt = 0; mt < 2; ++mt) {
#pragma unroll
    for (int nt = 0; nt < 4; ++nt) {
      int col = bn + wn + nt * 16 + lm;
      float bv = bias ? bias[col] : 0.0f;
#pragma unroll
      for (int j = 0; j < 8; ++j) {
        int row = bm + wm + mt * 16 + (lh ? j + 8 : j);
        float v = acc[mt][nt][j] + bv;
        if (relu) v = fmaxf(v, 0.0f);
        if (Cf) Cf[(size_t)row * ldc + col] = v;
        if (Ch) Ch[(size_t)row * ldc + col] = (__bf16)v;
      }
    }
  }
}

// ---------------------------------------------------------------------------
// Attention: one wave handles 16 query rows of one (batch, head).
// scores (WMMA) -> LDS -> softmax -> ctx = P*V (WMMA).
// ---------------------------------------------------------------------------
__global__ __launch_bounds__(32) void wmma_attention(
    const __bf16* __restrict__ Q,
    const __bf16* __restrict__ K,
    const __bf16* __restrict__ V,
    __bf16* __restrict__ ctx)
{
  __shared__ alignas(128) __bf16 P[16 * SEQ];    // 32 KB score/prob rows
  __shared__ alignas(128) __bf16 Vt[64 * BKP];   // transposed V k-tile

  const int lane = threadIdx.x;
  const int lm = lane & 15, lh = lane >> 4;
  const int blk = blockIdx.x;
  const int qt = blk & 63;          // query tile (16 rows)
  const int bh = blk >> 6;
  const int b = bh >> 4, h = bh & 15;
  const size_t rs = D_MODEL;                       // row stride (elements)
  const size_t base = ((size_t)b * SEQ) * rs + (size_t)h * D_K;

  // Q fragments (A operand, K = d_k = 64 -> two k=32 fragments)
  const __bf16* qrp = Q + base + (size_t)(qt * 16 + lm) * rs;
  v16bf qf0 = ld_frag(qrp + lh * 8,      qrp + lh * 8 + 16);
  v16bf qf1 = ld_frag(qrp + 32 + lh * 8, qrp + 32 + lh * 8 + 16);

  // Pass 1: scores = (Q K^T) / 8 into LDS (bf16)
  for (int kt = 0; kt < SEQ / 16; ++kt) {
    const __bf16* krp = K + base + (size_t)(kt * 16 + lm) * rs;
    v16bf kf0 = ld_frag(krp + lh * 8,      krp + lh * 8 + 16);
    v16bf kf1 = ld_frag(krp + 32 + lh * 8, krp + 32 + lh * 8 + 16);
    v8f s = {};
    s = WMMA_BF16(qf0, kf0, s);
    s = WMMA_BF16(qf1, kf1, s);
#pragma unroll
    for (int j = 0; j < 8; ++j) {
      int r = lh ? j + 8 : j;
      P[r * SEQ + kt * 16 + lm] = (__bf16)(s[j] * 0.125f);
    }
  }
  __syncthreads();

  // Softmax: two lanes per row, each scans 512 entries; combine via shfl_xor(16).
  {
    __bf16* pr = &P[lm * SEQ + lh * (SEQ / 2)];
    float mx = -3.0e38f;
    for (int i = 0; i < SEQ / 2; ++i) mx = fmaxf(mx, (float)pr[i]);
    mx = fmaxf(mx, __shfl_xor(mx, 16));
    float sum = 0.0f;
    for (int i = 0; i < SEQ / 2; ++i) sum += __expf((float)pr[i] - mx);
    sum += __shfl_xor(sum, 16);
    float inv = 1.0f / sum;
    for (int i = 0; i < SEQ / 2; ++i)
      pr[i] = (__bf16)(__expf((float)pr[i] - mx) * inv);
  }
  __syncthreads();

  // Pass 2: ctx = P * V  (M=16, N=64, K=1024)
  v8f o[4] = {};
  for (int kk = 0; kk < SEQ; kk += 32) {
    // stage V[kk..kk+31][0..63] transposed into Vt[n][k]
#pragma unroll
    for (int i = 0; i < 8; ++i) {
      int c  = lane + (i << 5);   // 0..255 x8 bf16
      int kr = c >> 3;            // 0..31
      int n8 = c & 7;
      v8bf row = *(const v8bf*)(V + base + (size_t)(kk + kr) * rs + n8 * 8);
#pragma unroll
      for (int e = 0; e < 8; ++e) Vt[(n8 * 8 + e) * BKP + kr] = row[e];
    }
    __syncthreads();
    const __bf16* pp = &P[lm * SEQ + kk + lh * 8];
    v16bf pf = ld_frag(pp, pp + 16);
#pragma unroll
    for (int nt = 0; nt < 4; ++nt) {
      const __bf16* vp = &Vt[(nt * 16 + lm) * BKP + lh * 8];
      o[nt] = WMMA_BF16(pf, ld_frag(vp, vp + 16), o[nt]);
    }
    __syncthreads();
  }

#pragma unroll
  for (int nt = 0; nt < 4; ++nt)
#pragma unroll
    for (int j = 0; j < 8; ++j) {
      int row = qt * 16 + (lh ? j + 8 : j);
      ctx[base + (size_t)row * rs + nt * 16 + lm] = (__bf16)o[nt][j];
    }
}

// ---------------------------------------------------------------------------
// out = LayerNorm(X + Y) * gamma + beta   (one block per token row, D=1024)
// ---------------------------------------------------------------------------
__global__ __launch_bounds__(256) void add_layernorm(
    const float* __restrict__ X, const float* __restrict__ Y,
    const float* __restrict__ gamma, const float* __restrict__ beta,
    float* __restrict__ outf, __bf16* __restrict__ outh)
{
  __shared__ float rsum[8], rsq[8];
  const int row = blockIdx.x;
  const int tid = threadIdx.x;
  const float* xr = X + (size_t)row * D_MODEL;
  const float* yr = Y + (size_t)row * D_MODEL;
  float v[4]; float s = 0.0f, sq = 0.0f;
#pragma unroll
  for (int i = 0; i < 4; ++i) {
    int c = tid + i * 256;
    v[i] = xr[c] + yr[c];
    s += v[i]; sq += v[i] * v[i];
  }
#pragma unroll
  for (int off = 16; off >= 1; off >>= 1) {
    s  += __shfl_xor(s, off);
    sq += __shfl_xor(sq, off);
  }
  if ((tid & 31) == 0) { rsum[tid >> 5] = s; rsq[tid >> 5] = sq; }
  __syncthreads();
  s = 0.0f; sq = 0.0f;
#pragma unroll
  for (int i = 0; i < 8; ++i) { s += rsum[i]; sq += rsq[i]; }
  const float mean = s * (1.0f / D_MODEL);
  const float var  = sq * (1.0f / D_MODEL) - mean * mean;
  const float rstd = rsqrtf(var + 1e-5f);
#pragma unroll
  for (int i = 0; i < 4; ++i) {
    int c = tid + i * 256;
    float o = (v[i] - mean) * rstd * gamma[c] + beta[c];
    if (outf) outf[(size_t)row * D_MODEL + c] = o;
    if (outh) outh[(size_t)row * D_MODEL + c] = (__bf16)o;
  }
}

__global__ void f32_to_bf16(const float* __restrict__ in, __bf16* __restrict__ out, int n) {
  int i = blockIdx.x * blockDim.x + threadIdx.x;
  if (i < n) out[i] = (__bf16)in[i];
}

// ---------------------------------------------------------------------------
// Host orchestration
// ---------------------------------------------------------------------------
extern "C" void kernel_launch(void* const* d_in, const int* in_sizes, int n_in,
                              void* d_out, int out_size, void* d_ws, size_t ws_size,
                              hipStream_t stream) {
  (void)in_sizes; (void)n_in; (void)out_size; (void)ws_size;
  const float* x   = (const float*)d_in[0];
  const float* Wq  = (const float*)d_in[1];
  const float* bq  = (const float*)d_in[2];
  const float* Wk  = (const float*)d_in[3];
  const float* bk  = (const float*)d_in[4];
  const float* Wv  = (const float*)d_in[5];
  const float* bv  = (const float*)d_in[6];
  const float* Wo  = (const float*)d_in[7];
  const float* bo  = (const float*)d_in[8];
  const float* g1  = (const float*)d_in[9];
  const float* b1  = (const float*)d_in[10];
  const float* W1  = (const float*)d_in[11];
  const float* bf1 = (const float*)d_in[12];
  const float* W2  = (const float*)d_in[13];
  const float* bf2 = (const float*)d_in[14];
  const float* g2  = (const float*)d_in[15];
  const float* b2  = (const float*)d_in[16];

  char* w = (char*)d_ws;
  auto alloc = [&](size_t bytes) -> void* {
    void* p = (void*)w;
    w += (bytes + 255) & ~(size_t)255;
    return p;
  };

  __bf16* xb   = (__bf16*)alloc((size_t)NTOK * D_MODEL * 2);
  __bf16* Wqb  = (__bf16*)alloc((size_t)D_MODEL * D_MODEL * 2);
  __bf16* Wkb  = (__bf16*)alloc((size_t)D_MODEL * D_MODEL * 2);
  __bf16* Wvb  = (__bf16*)alloc((size_t)D_MODEL * D_MODEL * 2);
  __bf16* Wob  = (__bf16*)alloc((size_t)D_MODEL * D_MODEL * 2);
  __bf16* W1b  = (__bf16*)alloc((size_t)D_MODEL * D_FF * 2);
  __bf16* W2b  = (__bf16*)alloc((size_t)D_FF * D_MODEL * 2);
  __bf16* Qb   = (__bf16*)alloc((size_t)NTOK * D_MODEL * 2);
  __bf16* Kb   = (__bf16*)alloc((size_t)NTOK * D_MODEL * 2);
  __bf16* Vb   = (__bf16*)alloc((size_t)NTOK * D_MODEL * 2);
  __bf16* ctxb = (__bf16*)alloc((size_t)NTOK * D_MODEL * 2);
  float*  attnf= (float*) alloc((size_t)NTOK * D_MODEL * 4);
  float*  hf   = (float*) alloc((size_t)NTOK * D_MODEL * 4);
  __bf16* hb   = (__bf16*)alloc((size_t)NTOK * D_MODEL * 2);
  __bf16* ff1b = (__bf16*)alloc((size_t)NTOK * D_FF * 2);
  float*  ff2f = (float*) alloc((size_t)NTOK * D_MODEL * 4);

  auto cvt = [&](const float* src, __bf16* dst, int n) {
    f32_to_bf16<<<(n + 255) / 256, 256, 0, stream>>>(src, dst, n);
  };
  cvt(x,  xb,  NTOK * D_MODEL);
  cvt(Wq, Wqb, D_MODEL * D_MODEL);
  cvt(Wk, Wkb, D_MODEL * D_MODEL);
  cvt(Wv, Wvb, D_MODEL * D_MODEL);
  cvt(Wo, Wob, D_MODEL * D_MODEL);
  cvt(W1, W1b, D_MODEL * D_FF);
  cvt(W2, W2b, D_FF * D_MODEL);

  const dim3 blk256(256);
  const dim3 gDD(D_MODEL / BN, NTOK / BM);   // (8, 32)
  const dim3 gDF(D_FF / BN,    NTOK / BM);   // (32, 32)

  // QKV projections (bf16 outputs feed attention)
  wmma_gemm_bf16<<<gDD, blk256, 0, stream>>>(xb, D_MODEL, Wqb, D_MODEL, bq,
                                             nullptr, Qb, D_MODEL, D_MODEL, 0);
  wmma_gemm_bf16<<<gDD, blk256, 0, stream>>>(xb, D_MODEL, Wkb, D_MODEL, bk,
                                             nullptr, Kb, D_MODEL, D_MODEL, 0);
  wmma_gemm_bf16<<<gDD, blk256, 0, stream>>>(xb, D_MODEL, Wvb, D_MODEL, bv,
                                             nullptr, Vb, D_MODEL, D_MODEL, 0);

  // Attention: 4 batches * 16 heads * 64 query tiles
  wmma_attention<<<4 * N_HEADS * (SEQ / 16), 32, 0, stream>>>(Qb, Kb, Vb, ctxb);

  // Output projection
  wmma_gemm_bf16<<<gDD, blk256, 0, stream>>>(ctxb, D_MODEL, Wob, D_MODEL, bo,
                                             attnf, nullptr, D_MODEL, D_MODEL, 0);

  // h = LN(x + attn_out)
  add_layernorm<<<NTOK, blk256, 0, stream>>>(x, attnf, g1, b1, hf, hb);

  // FFN
  wmma_gemm_bf16<<<gDF, blk256, 0, stream>>>(hb, D_MODEL, W1b, D_FF, bf1,
                                             nullptr, ff1b, D_FF, D_MODEL, 1);
  wmma_gemm_bf16<<<gDD, blk256, 0, stream>>>(ff1b, D_FF, W2b, D_MODEL, bf2,
                                             ff2f, nullptr, D_MODEL, D_FF, 0);

  // out = LN(h + ff)
  add_layernorm<<<NTOK, blk256, 0, stream>>>(hf, ff2f, g2, b2, (float*)d_out, nullptr);
}
